// NerveAttentionNetwork_12043088298053
// MI455X (gfx1250) — matrix operand order repaired and checked by reading, not compiled, not run on hardware
//
#include <hip/hip_runtime.h>
#include <math.h>

typedef __attribute__((ext_vector_type(16))) _Float16 v16h;
typedef __attribute__((ext_vector_type(8)))  _Float16 v8h;
typedef __attribute__((ext_vector_type(4)))  _Float16 v4h;
typedef __attribute__((ext_vector_type(8)))  float    v8f;

#define B_ 8
#define N_ 512
#define F_ 128
#define D_ 64
#define NEG_SLOPE 0.2f
#define JCHUNK 64

// ---- WMMA f16 16x16x32 fragment loaders (CDNA5 ISA §7.12.2) ----
// A (16x32, MxK), lane holds row M=lane&15:
//   lanes 0-15 : halves 0..7 -> K=0..7,   halves 8..15 -> K=16..23
//   lanes 16-31: halves 0..7 -> K=8..15,  halves 8..15 -> K=24..31
__device__ __forceinline__ v16h loadA16(const _Float16* rowk0, int lane) {
  const int ofs = (lane & 16) ? 8 : 0;
  union { v16h v; v8h h[2]; } u;
  u.h[0] = *(const v8h*)(rowk0 + ofs);
  u.h[1] = *(const v8h*)(rowk0 + ofs + 16);
  return u.v;
}
// B (32x16, KxN), lane holds col N=lane&15:
//   lanes 0-15 : halves 0..7 -> K=0..7,   halves 8..15 -> K=16..23
//   lanes 16-31: halves 0..7 -> K=8..15,  halves 8..15 -> K=24..31 (K-contiguous column).
__device__ __forceinline__ v16h loadB16(const _Float16* colk0, int lane) {
  const int ofs = (lane & 16) ? 8 : 0;
  union { v16h v; v8h h[2]; } u;
  u.h[0] = *(const v8h*)(colk0 + ofs);
  u.h[1] = *(const v8h*)(colk0 + ofs + 16);
  return u.v;
}

// ---------------------------------------------------------------------------
// xl = hin @ Wl, xr = hin @ Wr   (hin: [4096, Fin] f32, W: [Fin, 64] f32)
// Outputs: xl, xr (f32, node-major for scoring);
//          xl16T (f16, transposed per batch: [B][64][512]) for attention B op.
// ---------------------------------------------------------------------------
template <int FIN>
__global__ __launch_bounds__(128)
void gemm_xlxr(const float* __restrict__ hin,
               const float* __restrict__ Wl, const float* __restrict__ Wr,
               float* __restrict__ xl, float* __restrict__ xr,
               _Float16* __restrict__ xl16T)
{
  __shared__ __align__(16) _Float16 sA[16 * FIN];      // row-major 16 x FIN
  __shared__ __align__(16) _Float16 sWlT[D_ * FIN];    // transposed: [n][k]
  __shared__ __align__(16) _Float16 sWrT[D_ * FIN];

  const int tid = threadIdx.x;
  const int m0  = blockIdx.x * 16;

  for (int idx = tid; idx < 16 * (FIN / 4); idx += 128) {
    const int r  = idx / (FIN / 4);
    const int c4 = idx % (FIN / 4);
    const float4 v = ((const float4*)(hin + (size_t)(m0 + r) * FIN))[c4];
    v4h p; p[0] = (_Float16)v.x; p[1] = (_Float16)v.y;
           p[2] = (_Float16)v.z; p[3] = (_Float16)v.w;
    *(v4h*)&sA[r * FIN + c4 * 4] = p;
  }
  for (int idx = tid; idx < FIN * D_; idx += 128) {
    const int k = idx >> 6;
    const int n = idx & 63;
    sWlT[n * FIN + k] = (_Float16)Wl[idx];
    sWrT[n * FIN + k] = (_Float16)Wr[idx];
  }
  __syncthreads();

  const int wave = tid >> 5;
  const int lane = tid & 31;
  const int n0   = wave * 16;
  const int mrow = lane & 15;
  const int ncol = lane & 15;

  const _Float16* arow  = sA   + mrow * FIN;
  const _Float16* wlcol = sWlT + (n0 + ncol) * FIN;
  const _Float16* wrcol = sWrT + (n0 + ncol) * FIN;

  v8f accL = {};
  v8f accR = {};
#pragma unroll
  for (int k0 = 0; k0 < FIN; k0 += 32) {
    const v16h a  = loadA16(arow + k0, lane);
    const v16h bl = loadB16(wlcol + k0, lane);
    const v16h br = loadB16(wrcol + k0, lane);
    accL = __builtin_amdgcn_wmma_f32_16x16x32_f16(false, a, false, bl,
                                                  (short)0, accL, false, false);
    accR = __builtin_amdgcn_wmma_f32_16x16x32_f16(false, a, false, br,
                                                  (short)0, accR, false, false);
  }

  const int mofs = (lane & 16) ? 8 : 0;
#pragma unroll
  for (int r = 0; r < 8; ++r) {
    const size_t o = (size_t)(m0 + r + mofs) * D_ + n0 + ncol;
    xl[o] = accL[r];
    xr[o] = accR[r];
  }
  {
    const int bidx  = m0 / N_;
    const int node0 = (m0 % N_) + mofs;            // multiple of 8 -> 16B aligned
    _Float16* colp = xl16T + ((size_t)bidx * D_ + n0 + ncol) * N_ + node0;
    union { v8h v; _Float16 e[8]; } st;
#pragma unroll
    for (int r = 0; r < 8; ++r) st.e[r] = (_Float16)accL[r];
    *(v8h*)colp = st.v;
  }
}

// ---------------------------------------------------------------------------
// Attention for one (batch, 16-row i-tile):
//   e[i,j] = a . leaky_relu(xr_i + xl_j)   (f32 VALU, LDS-tiled)
//   alpha  = softmax_j(e) -> stored f16 in A-fragment layout (row-major LDS)
//   out    = tanh(alpha @ xl + bias)       (16 chained WMMAs, K = 512)
// ---------------------------------------------------------------------------
__global__ __launch_bounds__(256)
void attn(const float* __restrict__ xl, const float* __restrict__ xr,
          const _Float16* __restrict__ xl16T,
          const float* __restrict__ avec, const float* __restrict__ bias,
          float* __restrict__ hout)
{
  __shared__ __align__(16) float s_xr[16][D_];          // 4 KB
  __shared__ float s_a[D_];
  // s_xl (scoring stage) and s_alpha (f16 alpha, aggregation stage) overlay:
  // s_xl: 64 x 68 f32 = 17408 B; s_alpha: 16 x 512 f16 = 16384 B.
  __shared__ __align__(16) char s_ovl[JCHUNK * (D_ + 4) * sizeof(float)];
  __shared__ __align__(16) float s_e[16][N_];           // 32 KB

  float (*s_xl)[D_ + 4]  = (float (*)[D_ + 4])s_ovl;
  _Float16 (*s_alpha)[N_] = (_Float16 (*)[N_])s_ovl;

  const int tid = threadIdx.x;
  const int b   = blockIdx.x >> 5;
  const int i0  = (blockIdx.x & 31) * 16;
  const size_t base = (size_t)b * N_ * D_;

  for (int idx = tid; idx < 16 * (D_ / 4); idx += 256) {
    const int r = idx / (D_ / 4), c4 = idx % (D_ / 4);
    ((float4*)&s_xr[r][0])[c4] =
        ((const float4*)(xr + base + (size_t)(i0 + r) * D_))[c4];
  }
  if (tid < D_) s_a[tid] = avec[tid];
  __syncthreads();

  // ---- scoring over staged xl chunks ----
  for (int jc = 0; jc < N_; jc += JCHUNK) {
    for (int idx = tid; idx < JCHUNK * (D_ / 4); idx += 256) {
      const int j = idx / (D_ / 4), c4 = idx % (D_ / 4);
      ((float4*)&s_xl[j][0])[c4] =
          ((const float4*)(xl + base + (size_t)(jc + j) * D_))[c4];
    }
    __syncthreads();
    for (int p = tid; p < 16 * JCHUNK; p += 256) {
      const int i  = p >> 6;
      const int jj = p & (JCHUNK - 1);
      float acc = 0.f;
#pragma unroll 8
      for (int d = 0; d < D_; ++d) {
        const float z  = s_xr[i][d] + s_xl[jj][d];
        const float lr = z > 0.f ? z : NEG_SLOPE * z;
        acc = fmaf(s_a[d], lr, acc);
      }
      s_e[i][jc + jj] = acc;
    }
    __syncthreads();
  }

  // ---- softmax: 8 waves x 2 rows; writes alpha as f16 (overlays s_xl) ----
  const int wave = tid >> 5;
  const int lane = tid & 31;
  for (int rr = 0; rr < 2; ++rr) {
    const int i = wave * 2 + rr;
    float m = -1e30f;
    for (int j = lane; j < N_; j += 32) m = fmaxf(m, s_e[i][j]);
#pragma unroll
    for (int off = 16; off > 0; off >>= 1)
      m = fmaxf(m, __shfl_xor(m, off, 32));
    float s = 0.f;
    float ev[N_ / 32];
#pragma unroll 4
    for (int t = 0; t < N_ / 32; ++t) {
      ev[t] = __expf(s_e[i][lane + 32 * t] - m);
      s += ev[t];
    }
#pragma unroll
    for (int off = 16; off > 0; off >>= 1)
      s += __shfl_xor(s, off, 32);
    const float inv = 1.f / s;
#pragma unroll 4
    for (int t = 0; t < N_ / 32; ++t)
      s_alpha[i][lane + 32 * t] = (_Float16)(ev[t] * inv);
  }
  __syncthreads();

  // ---- aggregation: waves 0..3 own one 16-col N tile; K = 512 ----
  if (wave < 4) {
    const int n0   = wave * 16;
    const int mrow = lane & 15;
    const int ncol = lane & 15;
    const _Float16* colbase =
        xl16T + ((size_t)b * D_ + n0 + ncol) * N_;     // K-contiguous column

    v8f acc = {};
    v16h bb = loadB16(colbase, lane);                  // prefetch first B
#pragma unroll
    for (int k0 = 0; k0 < N_; k0 += 32) {
      v16h bnext = bb;
      if (k0 + 32 < N_) bnext = loadB16(colbase + k0 + 32, lane);
      const v16h aa = loadA16(&s_alpha[mrow][k0], lane);
      acc = __builtin_amdgcn_wmma_f32_16x16x32_f16(false, aa, false, bb,
                                                   (short)0, acc, false, false);
      bb = bnext;
    }
    const int mofs = (lane & 16) ? 8 : 0;
#pragma unroll
    for (int r = 0; r < 8; ++r) {
      const float v = acc[r] + bias[n0 + ncol];
      hout[base + (size_t)(i0 + r + mofs) * D_ + n0 + ncol] = tanhf(v);
    }
  }
}

// ---------------------------------------------------------------------------
// Mean-pool over nodes: [B, N, 64] -> [B, 64]
// ---------------------------------------------------------------------------
__global__ __launch_bounds__(64)
void pool_mean(const float* __restrict__ h, float* __restrict__ out)
{
  const int b = blockIdx.x;
  const int n = threadIdx.x;
  float s = 0.f;
  for (int j = 0; j < N_; ++j)
    s += h[((size_t)b * N_ + j) * D_ + n];
  out[b * D_ + n] = s * (1.0f / N_);
}

// ---------------------------------------------------------------------------
extern "C" void kernel_launch(void* const* d_in, const int* in_sizes, int n_in,
                              void* d_out, int out_size, void* d_ws, size_t ws_size,
                              hipStream_t stream)
{
  (void)in_sizes; (void)n_in; (void)out_size; (void)ws_size;

  const float* features = (const float*)d_in[0];
  float* out = (float*)d_out;

  // workspace (floats): xl | xr | xl16T (f16) | hbuf   (~3 MB total)
  const size_t NROWS = (size_t)B_ * N_;                 // 4096
  float*    xl    = (float*)d_ws;
  float*    xr    = xl + NROWS * D_;
  _Float16* xl16T = (_Float16*)(xr + NROWS * D_);       // [B][64][512] f16
  float*    hbuf  = (float*)(xl16T + NROWS * D_);

  const int mtiles = (int)(NROWS / 16);                 // 256
  const int atiles = B_ * (N_ / 16);                    // 256

  for (int br = 0; br < 2; ++br) {
    const int o = br * 8;
    const float* Wl1 = (const float*)d_in[1 + o];
    const float* Wr1 = (const float*)d_in[2 + o];
    const float* a1  = (const float*)d_in[3 + o];
    const float* b1  = (const float*)d_in[4 + o];
    const float* Wl2 = (const float*)d_in[5 + o];
    const float* Wr2 = (const float*)d_in[6 + o];
    const float* a2  = (const float*)d_in[7 + o];
    const float* b2  = (const float*)d_in[8 + o];

    gemm_xlxr<F_><<<mtiles, 128, 0, stream>>>(features, Wl1, Wr1, xl, xr, xl16T);
    attn<<<atiles, 256, 0, stream>>>(xl, xr, xl16T, a1, b1, hbuf);
    gemm_xlxr<D_><<<mtiles, 128, 0, stream>>>(hbuf, Wl2, Wr2, xl, xr, xl16T);
    attn<<<atiles, 256, 0, stream>>>(xl, xr, xl16T, a2, b2, hbuf);
    pool_mean<<<B_, 64, 0, stream>>>(hbuf, out + (size_t)br * B_ * D_);
  }
}
